// HMM_50422916055276
// MI455X (gfx1250) — compile-verified, matrix operand order checked by reading, not run on hardware
//
#include <hip/hip_runtime.h>

// HMM forward-backward, B=16, T=1024, K=1024.
// Each scan step: transfer(x)[b,j] = m_b + log( sum_i exp(x[b,i]-m_b) * A[i,j] )
// => 16x1024 @ 1024x1024 GEMM per step, done with v_wmma_f32_16x16x32_f16.
// FP8 would flush A's off-diagonal (eps/K = 9.77e-5 < E4M3 min denorm) to zero,
// so f16 inputs + f32 accumulation is the correct CDNA5 precision point.
//
// Round-2: kt-outer / N-tile-inner with 4 live accumulators (A-frag reuse).
// Round-3 fix: forbid unrolling of the kt loop -- full unroll made the
// compiler keep ~36 fragments live and spill them to scratch (= swizzled
// global memory on gfx1250). Rolled, the body is 2 ds_load + 4 global_load
// + 4 wmma with bounded registers; 4 waves/SIMD hide the remaining latency.

#define Bn 16
#define Tn 1024
#define Kn 1024

static constexpr float kHalfLn2Pi = 0.91893853320467274f; // 0.5*log(2*pi)

typedef _Float16 v8h  __attribute__((ext_vector_type(8)));
typedef _Float16 v16h __attribute__((ext_vector_type(16)));
typedef float    v8f  __attribute__((ext_vector_type(8)));

// LDS row strides padded so rows land on different bank groups.
#define FSTRIDE 1028   // floats per row of f  (4112 B/row, 16B aligned)
#define PSTRIDE 1032   // halves per row of p  (2064 B/row, 16B aligned)

// ---------------------------------------------------------------------------
// Pass 0: A = exp(ln_A) packed into WMMA B-fragment order, plus emission
// constants emul[k] = 0.5*exp(-2*ln_sigma), eoff[k] = -ln_sigma - 0.5*ln(2pi).
// Packed layout: flat = ((jt*32 + kt)*32 + lane)*16 + h
//   k = kt*32 + ((lane&16)?8:0) + (h<8 ? h : h+8),  j = jt*16 + (lane&15)
// matching the 16-bit A-operand k-split so one v16h load per lane per tile.
// ---------------------------------------------------------------------------
__global__ __launch_bounds__(256) void hmm_precompute(
    const float* __restrict__ lnA, const float* __restrict__ ln_sigma,
    _Float16* __restrict__ Apack, float* __restrict__ emul,
    float* __restrict__ eoff) {
  unsigned o    = blockIdx.x * 256u + threadIdx.x;     // Kn*Kn threads
  unsigned h    = o & 15u;
  unsigned lane = (o >> 4) & 31u;
  unsigned kt   = (o >> 9) & 31u;
  unsigned jt   = o >> 14;
  unsigned kb8  = (lane & 16u) ? 8u : 0u;
  unsigned i    = kt * 32u + kb8 + (h < 8u ? h : h + 8u);
  unsigned j    = jt * 16u + (lane & 15u);
  Apack[o] = (_Float16)__expf(lnA[i * Kn + j]);
  if (o < Kn) {
    float ls = ln_sigma[o];
    emul[o]  = 0.5f * __expf(-2.0f * ls);
    eoff[o]  = -ls - kHalfLn2Pi;
  }
}

// ---------------------------------------------------------------------------
// Pass 1: the two log-space scans. block 0 = forward, block 1 = backward.
// 512 threads = 16 waves; wave w owns state row b=w for max/exp and 4 N-tiles
// for the GEMM (32 kt-steps x 4 tiles = 128 WMMAs per wave per time step).
// Forward writes ln_f (emission included) directly into d_out [B,T,K];
// backward writes ln_b (pre-emission, faithful to the reference) into ws.
// ---------------------------------------------------------------------------
__global__ __launch_bounds__(512) void hmm_scan(
    const float* __restrict__ obvs, const float* __restrict__ mu,
    const float* __restrict__ ln_pi, const _Float16* __restrict__ Apack,
    const float* __restrict__ emul, const float* __restrict__ eoff,
    float* __restrict__ ln_f_out,   // = d_out, [B,T,K]
    float* __restrict__ ln_b_out) { // = ws,    [B,T,K]
  __shared__ float    f_s[Bn][FSTRIDE] __attribute__((aligned(16)));
  __shared__ _Float16 p_s[Bn][PSTRIDE] __attribute__((aligned(16)));
  __shared__ float    m_s[Bn];
  __shared__ float    obs_s[Bn];

  const int dir  = blockIdx.x;      // 0 = forward, 1 = backward
  const int tid  = threadIdx.x;
  const int lane = tid & 31;
  const int wv   = tid >> 5;        // wave id, 0..15

  float* gout = dir ? ln_b_out : ln_f_out;

  // ---- init at t0: x = ln_pi + ln_e(t0); store ln_pi (bwd) / x (fwd)
  const int t0 = dir ? (Tn - 1) : 0;
  for (int idx = tid; idx < Bn * Kn; idx += 512) {
    int b = idx >> 10, k = idx & (Kn - 1);
    float d  = obvs[b * Tn + t0] - mu[k];
    float e  = -(d * d) * emul[k] + eoff[k];
    float pi = ln_pi[k];
    f_s[b][k] = pi + e;
    gout[((size_t)b * Tn + t0) * Kn + k] = dir ? pi : (pi + e);
  }

  const int kb8  = (lane & 16) ? 8 : 0;
  const int hi8  = (lane & 16) ? 8 : 0;   // row offset for C/D layout
  const int mrow = lane & 15;
  const int jt0  = wv * 4;
  // B-fragment base for this wave's first tile; consecutive tiles are
  // TSTRIDE halves apart at fixed kt, consecutive kt are 512 halves apart.
  const size_t TSTRIDE = (size_t)32 * 32 * 16;  // 16384 halves = 32 KB
  const _Float16* __restrict__ bp =
      Apack + (((size_t)jt0 * 32) * 32 + lane) * 16;

  for (int s = 1; s < Tn; ++s) {
    const int tt = dir ? (Tn - 1 - s) : s;
    __syncthreads();  // previous step's f writes visible

    // ---- row max (wave wv owns row wv): shfl reduction over wave32
    float mx = -INFINITY;
    for (int k = lane; k < Kn; k += 32) mx = fmaxf(mx, f_s[wv][k]);
    for (int off = 16; off > 0; off >>= 1)
      mx = fmaxf(mx, __shfl_xor(mx, off, 32));
    if (lane == 0) m_s[wv] = mx;
    if (tid < Bn) obs_s[tid] = obvs[tid * Tn + tt];

    // ---- p = exp(f - m) in f16 (same wave, mx already broadcast)
    for (int k = lane; k < Kn; k += 32)
      p_s[wv][k] = (_Float16)__expf(f_s[wv][k] - mx);

    __syncthreads();  // p, m_s, obs_s ready for all waves

    // ---- GEMM: kt outer (rolled!), 4 N-tiles inner, A-frag reused 4x.
    v8f acc0 = {0.f,0.f,0.f,0.f,0.f,0.f,0.f,0.f};
    v8f acc1 = acc0, acc2 = acc0, acc3 = acc0;
#pragma clang loop unroll(disable)
    for (int kt = 0; kt < 32; ++kt) {
      v8h lo = *(const v8h*)&p_s[mrow][kt * 32 + kb8];
      v8h hi = *(const v8h*)&p_s[mrow][kt * 32 + kb8 + 16];
      v16h a = __builtin_shufflevector(lo, hi, 0, 1, 2, 3, 4, 5, 6, 7, 8, 9,
                                       10, 11, 12, 13, 14, 15);
      const _Float16* bk = bp + (size_t)kt * 512;
      v16h b0 = *(const v16h*)(bk);
      v16h b1 = *(const v16h*)(bk + TSTRIDE);
      v16h b2 = *(const v16h*)(bk + 2 * TSTRIDE);
      v16h b3 = *(const v16h*)(bk + 3 * TSTRIDE);
      acc0 = __builtin_amdgcn_wmma_f32_16x16x32_f16(false, a, false, b0,
                                                    (short)0, acc0, false, false);
      acc1 = __builtin_amdgcn_wmma_f32_16x16x32_f16(false, a, false, b1,
                                                    (short)0, acc1, false, false);
      acc2 = __builtin_amdgcn_wmma_f32_16x16x32_f16(false, a, false, b2,
                                                    (short)0, acc2, false, false);
      acc3 = __builtin_amdgcn_wmma_f32_16x16x32_f16(false, a, false, b3,
                                                    (short)0, acc3, false, false);
    }

    // ---- fused epilogue for the 4 tiles
    float m_r[8], obs_r[8];
#pragma unroll
    for (int r = 0; r < 8; ++r) {
      m_r[r]   = m_s[r + hi8];
      obs_r[r] = obs_s[r + hi8];
    }
#pragma unroll
    for (int jj = 0; jj < 4; ++jj) {
      v8f acc = (jj == 0) ? acc0 : (jj == 1) ? acc1 : (jj == 2) ? acc2 : acc3;
      const int j = (jt0 + jj) * 16 + (lane & 15);
      const float muj = mu[j], em = emul[j], eo = eoff[j];
#pragma unroll
      for (int r = 0; r < 8; ++r) {
        const int b  = r + hi8;
        float dd  = obs_r[r] - muj;
        float e   = -(dd * dd) * em + eo;
        float noE = m_r[r] + __logf(acc[r]);   // transfer(x)[b,j]
        float wE  = noE + e;                   // + emission at tt
        f_s[b][j] = wE;                        // state for next step
        gout[((size_t)b * Tn + tt) * Kn + j] = dir ? noE : wE;
      }
    }
  }
}

// ---------------------------------------------------------------------------
// Pass 2: gamma = (ln_f + ln_b) - logsumexp_k, in place on d_out.
// One 256-thread block per (b,t) row; pure bandwidth (~192 MB total).
// ---------------------------------------------------------------------------
__global__ __launch_bounds__(256) void hmm_gamma(
    float* __restrict__ out, const float* __restrict__ lnb) {
  __shared__ float red[8];
  const size_t base = (size_t)blockIdx.x * Kn;
  const int tid = threadIdx.x, lane = tid & 31, wv = tid >> 5;
  float v[4];
  float mx = -INFINITY;
#pragma unroll
  for (int i = 0; i < 4; ++i) {
    int k = tid + i * 256;
    v[i] = out[base + k] + lnb[base + k];
    mx = fmaxf(mx, v[i]);
  }
  for (int off = 16; off > 0; off >>= 1)
    mx = fmaxf(mx, __shfl_xor(mx, off, 32));
  if (lane == 0) red[wv] = mx;
  __syncthreads();
  mx = red[0];
#pragma unroll
  for (int w = 1; w < 8; ++w) mx = fmaxf(mx, red[w]);
  float sm = 0.f;
#pragma unroll
  for (int i = 0; i < 4; ++i) sm += __expf(v[i] - mx);
  for (int off = 16; off > 0; off >>= 1) sm += __shfl_xor(sm, off, 32);
  __syncthreads();  // everyone done reading red before reuse
  if (lane == 0) red[wv] = sm;
  __syncthreads();
  sm = 0.f;
#pragma unroll
  for (int w = 0; w < 8; ++w) sm += red[w];
  float lse = mx + __logf(sm);
#pragma unroll
  for (int i = 0; i < 4; ++i) out[base + tid + i * 256] = v[i] - lse;
}

// ---------------------------------------------------------------------------
extern "C" void kernel_launch(void* const* d_in, const int* in_sizes, int n_in,
                              void* d_out, int out_size, void* d_ws,
                              size_t ws_size, hipStream_t stream) {
  (void)in_sizes; (void)n_in; (void)out_size; (void)ws_size;
  const float* obvs     = (const float*)d_in[0];  // [B,T]
  const float* mu       = (const float*)d_in[1];  // [K]
  const float* ln_sigma = (const float*)d_in[2];  // [K]
  const float* ln_pi    = (const float*)d_in[3];  // [K]
  const float* lnA      = (const float*)d_in[4];  // [K,K]
  float* out = (float*)d_out;                     // [B,T,K]

  // workspace: ln_b [B,T,K] f32 (64MB) | Apack f16 (2MB) | emul,eoff (8KB)
  char*     ws    = (char*)d_ws;
  float*    lnb   = (float*)ws;
  _Float16* Apack = (_Float16*)(ws + (size_t)Bn * Tn * Kn * 4);
  float*    emul  = (float*)(ws + (size_t)Bn * Tn * Kn * 4 + (size_t)Kn * Kn * 2);
  float*    eoff  = emul + Kn;

  hipLaunchKernelGGL(hmm_precompute, dim3((Kn * Kn) / 256), dim3(256), 0,
                     stream, lnA, ln_sigma, Apack, emul, eoff);
  hipLaunchKernelGGL(hmm_scan, dim3(2), dim3(512), 0, stream, obvs, mu, ln_pi,
                     Apack, emul, eoff, out, lnb);
  hipLaunchKernelGGL(hmm_gamma, dim3(Bn * Tn), dim3(256), 0, stream, out, lnb);
}
